// MultiHeadAttentionAggregator_27736898798014
// MI455X (gfx1250) — matrix operand order repaired and checked by reading, not compiled, not run on hardware
//
#include <hip/hip_runtime.h>
#include <cstdint>

// Problem constants (from reference)
#define MM 400000
#define DD 256
#define AA 64
#define HH 8
#define NN 50000
#define NEG_SLOPE 0.2f

#define PITCH  264   // ushorts per LDS row for 256-wide tiles (256 + 8 pad) -> bank-offset 4 dwords/row
#define OPITCH 520   // ushorts per LDS row for 512-wide tiles (512 + 8 pad)

typedef __attribute__((ext_vector_type(16))) __bf16 v16bf;
typedef __attribute__((ext_vector_type(8)))  float  v8f;

union Frag {
  uint4 q[2];
  unsigned short u[16];
  v16bf v;
};

__device__ __forceinline__ unsigned short f2bf(float f) {
  unsigned u = __float_as_uint(f);
  u += 0x7FFFu + ((u >> 16) & 1u);   // round-to-nearest-even
  return (unsigned short)(u >> 16);
}

__device__ __forceinline__ v8f wmma_bf16(v16bf a, v16bf b, v8f c) {
  return __builtin_amdgcn_wmma_f32_16x16x32_bf16(false, a, false, b, (short)0, c, false, false);
}

// ---------------------------------------------------------------------------
// Init: seg_max = ordered-int(-inf), seg_sum = 0, out_h = 0
// ---------------------------------------------------------------------------
__global__ void k_init(int* __restrict__ segmax, float* __restrict__ segsum,
                       float* __restrict__ out_h) {
  int i = blockIdx.x * blockDim.x + threadIdx.x;
  if (i < NN * HH) { segmax[i] = (int)0x80000000; segsum[i] = 0.f; }
  size_t total  = (size_t)NN * (HH * AA);
  size_t stride = (size_t)gridDim.x * blockDim.x;
  for (size_t j = i; j < total; j += stride) out_h[j] = 0.f;
}

// ---------------------------------------------------------------------------
// Pack [H,D,A] f32 weights into bf16 B-fragment order:
//   id = ((h*4 + t)*8 + kk)*32 + lane ; element j -> W[h][kk*32 + koffB + (j<8 ? j : j+8)][t*16 + (lane&15)]
// ---------------------------------------------------------------------------
__global__ void k_pack_whda(const float* __restrict__ W, unsigned short* __restrict__ P) {
  int id = blockIdx.x * blockDim.x + threadIdx.x;
  if (id >= HH * 4 * 8 * 32) return;
  int lane = id & 31;
  int kk   = (id >> 5) & 7;
  int t    = (id >> 8) & 3;
  int h    = id >> 10;
  int n     = t * 16 + (lane & 15);
  int koffB = (lane < 16) ? 0 : 8;
  unsigned short* dst = P + (size_t)id * 16;
#pragma unroll
  for (int j = 0; j < 16; ++j) {
    int k = kk * 32 + koffB + (j < 8 ? j : j + 8);
    dst[j] = f2bf(W[((size_t)h * DD + k) * AA + n]);
  }
}

// Pack Wout [512,256] into bf16 B-fragment order:
//   id = (ct*16 + kk)*32 + lane ; element j -> Wout[kk*32 + koffB + (j<8 ? j : j+8)][ct*16 + (lane&15)]
__global__ void k_pack_wout(const float* __restrict__ W, unsigned short* __restrict__ P) {
  int id = blockIdx.x * blockDim.x + threadIdx.x;
  if (id >= 16 * 16 * 32) return;
  int lane = id & 31;
  int kk   = (id >> 5) & 15;
  int ct   = id >> 9;
  int n     = ct * 16 + (lane & 15);
  int koffB = (lane < 16) ? 0 : 8;
  unsigned short* dst = P + (size_t)id * 16;
#pragma unroll
  for (int j = 0; j < 16; ++j) {
    int k = kk * 32 + koffB + (j < 8 ? j : j + 8);
    dst[j] = f2bf(W[(size_t)k * DD + n]);
  }
}

// ---------------------------------------------------------------------------
// Pass 1: fused Q/K projection + per-row score + leaky ReLU + segment max
// grid = MM/16 blocks, 256 threads = 8 waves (one head each)
// ---------------------------------------------------------------------------
__global__ __launch_bounds__(256) void k_proj_score(
    const float* __restrict__ msg, const int* __restrict__ index,
    const unsigned short* __restrict__ Wq_p, const unsigned short* __restrict__ Wk_p,
    float* __restrict__ scores, int* __restrict__ segmax) {
  __shared__ __align__(16) unsigned short smsg[16 * PITCH];
  const int tile = blockIdx.x;
  const int row0 = tile * 16;
  const int tid  = threadIdx.x;

  { // stage 16x256 f32 -> bf16 LDS
    int r = tid >> 4, cs = (tid & 15) * 16;
    const float* src = msg + (size_t)(row0 + r) * DD + cs;
    unsigned short* dst = &smsg[r * PITCH + cs];
#pragma unroll
    for (int j = 0; j < 16; ++j) dst[j] = f2bf(src[j]);
  }
  __syncthreads();

  const int h    = tid >> 5;   // wave == head
  const int lane = tid & 31;
  const int r    = lane & 15;
  const int koff = (lane < 16) ? 0 : 8;
  const unsigned short* rowp = &smsg[r * PITCH];

  v8f qacc[4], kacc[4];
#pragma unroll
  for (int t = 0; t < 4; ++t) {
    qacc[t] = (v8f){0, 0, 0, 0, 0, 0, 0, 0};
    kacc[t] = (v8f){0, 0, 0, 0, 0, 0, 0, 0};
  }

#pragma unroll
  for (int kk = 0; kk < 8; ++kk) {
    Frag a;
    const uint4* ap = reinterpret_cast<const uint4*>(rowp + kk * 32 + koff);
    a.q[0] = ap[0];   // K = kk*32 + koff + 0..7
    a.q[1] = ap[2];   // K = kk*32 + 16 + koff + 0..7
#pragma unroll
    for (int t = 0; t < 4; ++t) {
      size_t off = ((((size_t)h * 4 + t) * 8 + kk) * 32 + lane) * 16;
      Frag bq, bk;
      const uint4* pq = reinterpret_cast<const uint4*>(Wq_p + off);
      bq.q[0] = pq[0]; bq.q[1] = pq[1];
      const uint4* pk = reinterpret_cast<const uint4*>(Wk_p + off);
      bk.q[0] = pk[0]; bk.q[1] = pk[1];
      qacc[t] = wmma_bf16(a.v, bq.v, qacc[t]);
      kacc[t] = wmma_bf16(a.v, bk.v, kacc[t]);
    }
  }

  // score[row] = sum_a q*k : lanewise product then 16-lane butterfly reduce
  float ps[8];
#pragma unroll
  for (int g = 0; g < 8; ++g) {
    float s = 0.f;
#pragma unroll
    for (int t = 0; t < 4; ++t) s += qacc[t][g] * kacc[t][g];
    s += __shfl_xor(s, 1, 32);
    s += __shfl_xor(s, 2, 32);
    s += __shfl_xor(s, 4, 32);
    s += __shfl_xor(s, 8, 32);
    ps[g] = s;
  }

  if ((lane & 15) == 0) {
    int rbase = (lane < 16) ? 0 : 8;
#pragma unroll
    for (int g = 0; g < 8; ++g) {
      float s = ps[g];
      s = (s >= 0.f) ? s : NEG_SLOPE * s;   // leaky relu
      int m = row0 + rbase + g;
      scores[(size_t)m * HH + h] = s;
      int idx = index[m];
      int bi  = __float_as_int(s);
      int enc = (bi >= 0) ? bi : (bi ^ 0x7FFFFFFF);  // ordered-int encoding
      atomicMax(&segmax[idx * HH + h], enc);
    }
  }
}

// ---------------------------------------------------------------------------
// Finalize seg_max: ordered int -> float in place; empty segment -> 0
// ---------------------------------------------------------------------------
__global__ void k_finalize_max(int* __restrict__ segmax) {
  int i = blockIdx.x * blockDim.x + threadIdx.x;
  if (i >= NN * HH) return;
  int v = segmax[i];
  float f;
  if (v == (int)0x80000000) f = 0.f;
  else {
    int d = (v >= 0) ? v : (v ^ 0x7FFFFFFF);
    f = __int_as_float(d);
  }
  reinterpret_cast<float*>(segmax)[i] = f;
}

// ---------------------------------------------------------------------------
// Pass 2: seg_sum += exp(score - seg_max)
// ---------------------------------------------------------------------------
__global__ void k_expsum(const float* __restrict__ scores, const int* __restrict__ index,
                         const float* __restrict__ segmaxf, float* __restrict__ segsum) {
  int i = blockIdx.x * blockDim.x + threadIdx.x;
  if (i >= MM * HH) return;
  int m = i >> 3, h = i & 7;
  int idx = index[m];
  float e = __expf(scores[i] - segmaxf[idx * HH + h]);
  atomicAdd(&segsum[idx * HH + h], e);
}

// ---------------------------------------------------------------------------
// Pass 3: recompute V via WMMA, scale by attn, scatter-add into out_h[N,H,A]
// grid = MM/16 blocks, 256 threads = 8 waves (one head each)
// ---------------------------------------------------------------------------
__global__ __launch_bounds__(256) void k_aggregate(
    const float* __restrict__ msg, const int* __restrict__ index,
    const unsigned short* __restrict__ Wv_p,
    const float* __restrict__ scores, const float* __restrict__ segmaxf,
    const float* __restrict__ segsum, float* __restrict__ out_h) {
  __shared__ __align__(16) unsigned short smsg[16 * PITCH];
  __shared__ float sattn[HH][16];
  __shared__ int   sidx[16];
  const int tile = blockIdx.x;
  const int row0 = tile * 16;
  const int tid  = threadIdx.x;

  { // stage msg tile
    int r = tid >> 4, cs = (tid & 15) * 16;
    const float* src = msg + (size_t)(row0 + r) * DD + cs;
    unsigned short* dst = &smsg[r * PITCH + cs];
#pragma unroll
    for (int j = 0; j < 16; ++j) dst[j] = f2bf(src[j]);
  }
  if (tid < 16) sidx[tid] = index[row0 + tid];
  __syncthreads();

  const int h    = tid >> 5;
  const int lane = tid & 31;

  if (lane < 16) { // attn for the 16 rows, this head
    int m   = row0 + lane;
    int idx = sidx[lane];
    float s  = scores[(size_t)m * HH + h];
    float mx = segmaxf[idx * HH + h];
    float sm = segsum[idx * HH + h];
    float e  = __expf(s - mx);
    sattn[h][lane] = (sm > 0.f) ? (e / sm) : 0.f;
  }
  __syncthreads();

  const int r    = lane & 15;
  const int koff = (lane < 16) ? 0 : 8;
  const unsigned short* rowp = &smsg[r * PITCH];

  v8f vacc[4];
#pragma unroll
  for (int t = 0; t < 4; ++t) vacc[t] = (v8f){0, 0, 0, 0, 0, 0, 0, 0};

#pragma unroll
  for (int kk = 0; kk < 8; ++kk) {
    Frag a;
    const uint4* ap = reinterpret_cast<const uint4*>(rowp + kk * 32 + koff);
    a.q[0] = ap[0];
    a.q[1] = ap[2];
#pragma unroll
    for (int t = 0; t < 4; ++t) {
      size_t off = ((((size_t)h * 4 + t) * 8 + kk) * 32 + lane) * 16;
      Frag bv;
      const uint4* pv = reinterpret_cast<const uint4*>(Wv_p + off);
      bv.q[0] = pv[0]; bv.q[1] = pv[1];
      vacc[t] = wmma_bf16(a.v, bv.v, vacc[t]);
    }
  }

  const int c0    = lane & 15;
  const int rbase = (lane < 16) ? 0 : 8;
#pragma unroll
  for (int t = 0; t < 4; ++t) {
    int col = t * 16 + c0;   // a in 0..63
#pragma unroll
    for (int g = 0; g < 8; ++g) {
      int rloc = rbase + g;
      float val = vacc[t][g] * sattn[h][rloc];
      int idx = sidx[rloc];
      atomicAdd(&out_h[((size_t)idx * HH + h) * AA + col], val);
    }
  }
}

// ---------------------------------------------------------------------------
// Pass 4: out = out_h @ Wout + bout   (N x 512) x (512 x 256)
// grid = NN/16 blocks, 256 threads = 8 waves, each wave does 2 column tiles
// ---------------------------------------------------------------------------
__global__ __launch_bounds__(256) void k_out_gemm(
    const float* __restrict__ out_h, const unsigned short* __restrict__ Wout_p,
    const float* __restrict__ bout, float* __restrict__ out) {
  __shared__ __align__(16) unsigned short sA[16 * OPITCH];
  const int tile = blockIdx.x;
  const int row0 = tile * 16;
  const int tid  = threadIdx.x;

  { // stage 16x512 f32 -> bf16 LDS
    int r = tid >> 4, cs = (tid & 15) * 32;
    const float* src = out_h + (size_t)(row0 + r) * (HH * AA) + cs;
    unsigned short* dst = &sA[r * OPITCH + cs];
#pragma unroll
    for (int j = 0; j < 32; ++j) dst[j] = f2bf(src[j]);
  }
  __syncthreads();

  const int wave = tid >> 5;
  const int lane = tid & 31;
  const int r    = lane & 15;
  const int koff = (lane < 16) ? 0 : 8;
  const unsigned short* rowp = &sA[r * OPITCH];

  for (int ct = wave; ct < 16; ct += 8) {
    v8f acc = (v8f){0, 0, 0, 0, 0, 0, 0, 0};
#pragma unroll
    for (int kk = 0; kk < 16; ++kk) {
      Frag a;
      const uint4* ap = reinterpret_cast<const uint4*>(rowp + kk * 32 + koff);
      a.q[0] = ap[0];
      a.q[1] = ap[2];
      size_t off = (((size_t)ct * 16 + kk) * 32 + lane) * 16;
      Frag b;
      const uint4* bp = reinterpret_cast<const uint4*>(Wout_p + off);
      b.q[0] = bp[0]; b.q[1] = bp[1];
      acc = wmma_bf16(a.v, b.v, acc);
    }
    int col  = ct * 16 + (lane & 15);
    float bi = bout[col];
    int rb   = (lane < 16) ? 0 : 8;
#pragma unroll
    for (int g = 0; g < 8; ++g)
      out[(size_t)(row0 + rb + g) * DD + col] = acc[g] + bi;
  }
}

// ---------------------------------------------------------------------------
// Host launch
// ---------------------------------------------------------------------------
static inline size_t align256(size_t x) { return (x + 255) & ~(size_t)255; }

extern "C" void kernel_launch(void* const* d_in, const int* in_sizes, int n_in,
                              void* d_out, int out_size, void* d_ws, size_t ws_size,
                              hipStream_t stream) {
  (void)in_sizes; (void)n_in; (void)out_size; (void)ws_size;
  const float* msg   = (const float*)d_in[0];
  const int*   index = (const int*)d_in[1];
  // d_in[2] = t (unused), d_in[3] = dim_size (compile-time NN)
  const float* Wq    = (const float*)d_in[4];
  const float* Wk    = (const float*)d_in[5];
  const float* Wv    = (const float*)d_in[6];
  const float* Wout  = (const float*)d_in[7];
  const float* bout  = (const float*)d_in[8];
  float* out = (float*)d_out;

  // workspace carve-up
  char* ws = (char*)d_ws;
  size_t o = 0;
  float* scores  = (float*)(ws + o); o += align256((size_t)MM * HH * 4);
  int*   segmax  = (int*)  (ws + o); o += align256((size_t)NN * HH * 4);
  float* segsum  = (float*)(ws + o); o += align256((size_t)NN * HH * 4);
  float* out_h   = (float*)(ws + o); o += align256((size_t)NN * HH * AA * 4);
  unsigned short* Wq_p   = (unsigned short*)(ws + o); o += align256((size_t)HH * DD * AA * 2);
  unsigned short* Wk_p   = (unsigned short*)(ws + o); o += align256((size_t)HH * DD * AA * 2);
  unsigned short* Wv_p   = (unsigned short*)(ws + o); o += align256((size_t)HH * DD * AA * 2);
  unsigned short* Wout_p = (unsigned short*)(ws + o); o += align256((size_t)HH * AA * DD * 2);

  // 0) init accumulators
  k_init<<<(NN * HH + 255) / 256, 256, 0, stream>>>(segmax, segsum, out_h);

  // 1) pack weights into bf16 fragment order
  k_pack_whda<<<(HH * 4 * 8 * 32 + 255) / 256, 256, 0, stream>>>(Wq, Wq_p);
  k_pack_whda<<<(HH * 4 * 8 * 32 + 255) / 256, 256, 0, stream>>>(Wk, Wk_p);
  k_pack_whda<<<(HH * 4 * 8 * 32 + 255) / 256, 256, 0, stream>>>(Wv, Wv_p);
  k_pack_wout<<<(16 * 16 * 32 + 255) / 256, 256, 0, stream>>>(Wout, Wout_p);

  // 2) fused Q/K projection + scores + segment max (WMMA)
  k_proj_score<<<MM / 16, 256, 0, stream>>>(msg, index, Wq_p, Wk_p, scores, segmax);

  // 3) finalize max, 4) exp-sum
  k_finalize_max<<<(NN * HH + 255) / 256, 256, 0, stream>>>(segmax);
  k_expsum<<<(MM * HH + 255) / 256, 256, 0, stream>>>(scores, index,
                                                      (const float*)segmax, segsum);

  // 5) V projection + weighted scatter (WMMA + atomics)
  k_aggregate<<<MM / 16, 256, 0, stream>>>(msg, index, Wv_p, scores,
                                           (const float*)segmax, segsum, out_h);

  // 6) output GEMM + bias (WMMA)
  k_out_gemm<<<NN / 16, 256, 0, stream>>>(out_h, Wout_p, bout, out);
}
